// TopKSparseCoder_49538152792954
// MI455X (gfx1250) — matrix operand order repaired and checked by reading, not compiled, not run on hardware
//
#include <hip/hip_runtime.h>
#include <hip/hip_bf16.h>
#include <math.h>

// ---------------------------------------------------------------------------
// FISTA top-k sparse coder for MI455X (gfx1250, wave32).
//   grad = Y@G - C with G = Dn Dn^T (1024x1024, f32->bf16 frags),
//   C = x@Dn^T (f32). Hot loop: one bf16 WMMA GEMM (f32 accum) fused with
//   V = Y - step*(Y@G - C), then per-row radix top-64 + momentum update.
//   Whole working set (~130 MB) is L2-resident (192 MB).
// ---------------------------------------------------------------------------

typedef __attribute__((ext_vector_type(16))) __bf16 v16bf;
typedef __attribute__((ext_vector_type(8)))  float  v8f;

#define B_ROWS   8192
#define N_ATOMS  1024
#define FDIM     128
#define KSPARSE  64
#define N_ITERS  30
#define N_POWER  20

// ---------------- normalize dictionary rows --------------------------------
__global__ void k_norm_D(const float* __restrict__ D, float* __restrict__ Dn) {
  const int row = blockIdx.x;
  const int t   = threadIdx.x;            // 128 threads, 1 elem each
  __shared__ float red[FDIM];
  float d = D[row * FDIM + t];
  red[t] = d * d;
  __syncthreads();
  for (int off = FDIM / 2; off > 0; off >>= 1) {
    if (t < off) red[t] += red[t + off];
    __syncthreads();
  }
  const float inv = 1.0f / (sqrtf(red[0]) + 1e-8f);
  Dn[row * FDIM + t] = d * inv;
}

// ---------------- G = Dn Dn^T (f32, one-time, 268 MFLOP) -------------------
__global__ void k_gram(const float* __restrict__ Dn, float* __restrict__ G) {
  const int j = blockIdx.x * 16 + threadIdx.x;
  const int i = blockIdx.y * 16 + threadIdx.y;
  const float4* a = (const float4*)(Dn + (size_t)i * FDIM);
  const float4* b = (const float4*)(Dn + (size_t)j * FDIM);
  float s = 0.0f;
#pragma unroll
  for (int k = 0; k < FDIM / 4; ++k) {
    float4 u = a[k], w = b[k];
    s += u.x * w.x + u.y * w.y + u.z * w.z + u.w * w.w;
  }
  G[(size_t)i * N_ATOMS + j] = s;
}

// ---------------- power iteration on G: step = 1/||G||_2 -------------------
__global__ void k_power(const float* __restrict__ G, float* __restrict__ stepOut) {
  __shared__ float v[N_ATOMS];
  __shared__ float red[N_ATOMS];
  const int t = threadIdx.x;              // 1024 threads

  unsigned int h = (unsigned int)t * 1103515245u + 12345u;
  h ^= h >> 13; h *= 2654435761u; h ^= h >> 16;
  v[t] = (float)(h & 0xFFFFu) * (1.0f / 65536.0f) + 0.0625f;  // generic non-orthogonal init
  __syncthreads();

  // normalize init
  red[t] = v[t] * v[t];
  __syncthreads();
  for (int off = 512; off > 0; off >>= 1) { if (t < off) red[t] += red[t + off]; __syncthreads(); }
  v[t] = v[t] / (sqrtf(red[0]) + 1e-12f);
  __syncthreads();

  float w = 0.0f;
  for (int it = 0; it < N_POWER; ++it) {
    const float4* gr = (const float4*)(G + (size_t)t * N_ATOMS);
    w = 0.0f;
    for (int k = 0; k < N_ATOMS / 4; ++k) {
      float4 g = gr[k];
      w += g.x * v[4 * k] + g.y * v[4 * k + 1] + g.z * v[4 * k + 2] + g.w * v[4 * k + 3];
    }
    red[t] = w * w;
    __syncthreads();
    for (int off = 512; off > 0; off >>= 1) { if (t < off) red[t] += red[t + off]; __syncthreads(); }
    const float inv = 1.0f / (sqrtf(red[0]) + 1e-12f);
    __syncthreads();
    v[t] = w * inv;
    __syncthreads();
  }
  // L = v^T G v
  const float4* gr = (const float4*)(G + (size_t)t * N_ATOMS);
  w = 0.0f;
  for (int k = 0; k < N_ATOMS / 4; ++k) {
    float4 g = gr[k];
    w += g.x * v[4 * k] + g.y * v[4 * k + 1] + g.z * v[4 * k + 2] + g.w * v[4 * k + 3];
  }
  red[t] = v[t] * w;
  __syncthreads();
  for (int off = 512; off > 0; off >>= 1) { if (t < off) red[t] += red[t + off]; __syncthreads(); }
  if (t == 0) stepOut[0] = 1.0f / fmaxf(red[0], 1e-8f);
}

// -------- pack G into WMMA B-fragment layout (bf16), ISA 7.12.2 ------------
// frag f = ntile*32 + kc; element (f, lane, j):
//   Gp[(f*32+lane)*16 + j] = G[kc*32 + (lane>>4)*8 + (j&7) + (j>>3)*16][ntile*16 + (lane&15)]
__global__ void k_pack_G(const float* __restrict__ G, __bf16* __restrict__ Gp) {
  const int idx  = blockIdx.x * 256 + threadIdx.x;   // 0 .. 1M-1
  const int j    = idx & 15;
  const int lane = (idx >> 4) & 31;
  const int f    = idx >> 9;
  const int kc    = f & 31;
  const int ntile = f >> 5;
  const int krow = kc * 32 + ((lane >> 4) << 3) + (j & 7) + ((j >> 3) << 4);
  const int col  = ntile * 16 + (lane & 15);
  Gp[idx] = (__bf16)G[(size_t)krow * N_ATOMS + col];
}

// ---------------- C = x @ Dn^T (f32, one-time, 2.1 GFLOP) ------------------
__global__ void k_xDT(const float* __restrict__ x, const float* __restrict__ Dn,
                      float* __restrict__ C) {
  const int j = blockIdx.x * 16 + threadIdx.x;   // atom
  const int m = blockIdx.y * 16 + threadIdx.y;   // batch row
  const float4* a = (const float4*)(x + (size_t)m * FDIM);
  const float4* b = (const float4*)(Dn + (size_t)j * FDIM);
  float s = 0.0f;
#pragma unroll
  for (int k = 0; k < FDIM / 4; ++k) {
    float4 u = a[k], w = b[k];
    s += u.x * w.x + u.y * w.y + u.z * w.z + u.w * w.w;
  }
  C[(size_t)m * N_ATOMS + j] = s;
}

__global__ void k_zero2(float* __restrict__ a, float* __restrict__ b) {
  const size_t i = (size_t)blockIdx.x * 256 + threadIdx.x;
  a[i] = 0.0f;
  b[i] = 0.0f;
}

// ---------------- hot loop GEMM: V = Y - step*(Y@G - C) --------------------
// 256 threads = 8 waves; wave -> 16x64 tile (4 accumulators); block -> 32x256.
__device__ inline v16bf cvt16(float4 f0, float4 f1, float4 f2, float4 f3) {
  v16bf a;
  a[0] = (__bf16)f0.x;  a[1] = (__bf16)f0.y;  a[2] = (__bf16)f0.z;  a[3] = (__bf16)f0.w;
  a[4] = (__bf16)f1.x;  a[5] = (__bf16)f1.y;  a[6] = (__bf16)f1.z;  a[7] = (__bf16)f1.w;
  a[8] = (__bf16)f2.x;  a[9] = (__bf16)f2.y;  a[10] = (__bf16)f2.z; a[11] = (__bf16)f2.w;
  a[12] = (__bf16)f3.x; a[13] = (__bf16)f3.y; a[14] = (__bf16)f3.z; a[15] = (__bf16)f3.w;
  return a;
}

__global__ __launch_bounds__(256)
void k_fista_gemm(const float* __restrict__ Yf, const __bf16* __restrict__ Gp,
                  const float* __restrict__ C, const float* __restrict__ stepPtr,
                  float* __restrict__ V) {
  const int lane    = threadIdx.x & 31;
  const int wave    = threadIdx.x >> 5;
  const int waveRow = wave >> 2;                 // 0..1
  const int waveCol = wave & 3;                  // 0..3
  const int rowBase  = blockIdx.y * 32 + waveRow * 16;
  const int colTile0 = blockIdx.x * 16 + waveCol * 4;   // units of 16 cols
  const int halfSel  = lane >> 4;                // 0/1
  const int laneMod  = lane & 15;

  // A-frag source: lane holds row (lane&15), K halves {kb..kb+7, kb+16..kb+23}
  const float* yrow = Yf + (size_t)(rowBase + laneMod) * N_ATOMS + (halfSel << 3);
  const v16bf* gbase = ((const v16bf*)Gp) + lane;

  v8f acc[4] = {v8f{}, v8f{}, v8f{}, v8f{}};

  for (int kc = 0; kc < 32; ++kc) {
    const float* yp = yrow + kc * 32;
    float4 f0 = *(const float4*)(yp + 0);
    float4 f1 = *(const float4*)(yp + 4);
    float4 f2 = *(const float4*)(yp + 16);
    float4 f3 = *(const float4*)(yp + 20);
    __builtin_prefetch((const void*)(yp + 32), 0, 1);       // global_prefetch_b8
    const v16bf a = cvt16(f0, f1, f2, f3);

    const v16bf b0 = gbase[(size_t)((colTile0 + 0) * 32 + kc) * 32];
    const v16bf b1 = gbase[(size_t)((colTile0 + 1) * 32 + kc) * 32];
    const v16bf b2 = gbase[(size_t)((colTile0 + 2) * 32 + kc) * 32];
    const v16bf b3 = gbase[(size_t)((colTile0 + 3) * 32 + kc) * 32];

    acc[0] = __builtin_amdgcn_wmma_f32_16x16x32_bf16(false, a, false, b0, (short)0, acc[0], false, false);
    acc[1] = __builtin_amdgcn_wmma_f32_16x16x32_bf16(false, a, false, b1, (short)0, acc[1], false, false);
    acc[2] = __builtin_amdgcn_wmma_f32_16x16x32_bf16(false, a, false, b2, (short)0, acc[2], false, false);
    acc[3] = __builtin_amdgcn_wmma_f32_16x16x32_bf16(false, a, false, b3, (short)0, acc[3], false, false);
  }

  const float step = stepPtr[0];
#pragma unroll
  for (int nt = 0; nt < 4; ++nt) {
    const int n = (colTile0 + nt) * 16 + laneMod;
#pragma unroll
    for (int r = 0; r < 8; ++r) {
      const int m = rowBase + (halfSel << 3) + r;    // C/D layout: M=(lane>=16)*8+r
      const size_t idx = (size_t)m * N_ATOMS + n;
      V[idx] = Yf[idx] - step * (acc[nt][r] - C[idx]);
    }
  }
}

// ---------------- per-row exact radix top-64 + FISTA update ----------------
__global__ __launch_bounds__(256)
void k_topk(const float* __restrict__ V, float* __restrict__ Z,
            float* __restrict__ Yf, float mom) {
  const int row = blockIdx.x;
  const int t   = threadIdx.x;                 // 256 threads, 4 elems each
  __shared__ unsigned int keys[N_ATOMS];
  __shared__ float        vals[N_ATOMS];
  __shared__ unsigned int hist[256];
  __shared__ unsigned int sh_digit;

  const size_t base = (size_t)row * N_ATOMS;
#pragma unroll
  for (int e = 0; e < 4; ++e) {
    const int c = e * 256 + t;
    const float v = V[base + c];
    vals[c] = v;
    keys[c] = __float_as_uint(v) & 0x7fffffffu;   // |v| as monotone key
  }
  __syncthreads();

  unsigned int prefix = 0u, pmask = 0u, remk = KSPARSE;
  for (int pass = 0; pass < 4; ++pass) {
    const int shift = 24 - 8 * pass;
    hist[t] = 0u;
    __syncthreads();
#pragma unroll
    for (int e = 0; e < 4; ++e) {
      const unsigned int k = keys[e * 256 + t];
      if ((k & pmask) == prefix) atomicAdd(&hist[(k >> shift) & 255u], 1u);
    }
    __syncthreads();
    // inclusive suffix scan: hist[d] = #keys with digit >= d (within candidates)
    for (int off = 1; off < 256; off <<= 1) {
      const unsigned int add = (t + off < 256) ? hist[t + off] : 0u;
      __syncthreads();
      hist[t] += add;
      __syncthreads();
    }
    const unsigned int cntGE = hist[t];
    const unsigned int cntGT = (t < 255) ? hist[t + 1] : 0u;
    if (cntGE >= remk && cntGT < remk) sh_digit = (unsigned int)t;  // unique crossing
    __syncthreads();
    const unsigned int d  = sh_digit;
    const unsigned int gt = (d < 255u) ? hist[d + 1] : 0u;
    remk  -= gt;
    prefix |= d << shift;
    pmask  |= 0xFFu << shift;
    __syncthreads();
  }
  const unsigned int T = prefix;               // exact 64th-largest |v| key

#pragma unroll
  for (int e = 0; e < 4; ++e) {
    const int c = e * 256 + t;
    const float v = vals[c];
    const float z = (keys[c] >= T) ? v : 0.0f;
    const float zo = Z[base + c];
    Z[base + c]  = z;
    Yf[base + c] = z + mom * (z - zo);
  }
}

// ---------------------------------------------------------------------------
extern "C" void kernel_launch(void* const* d_in, const int* in_sizes, int n_in,
                              void* d_out, int out_size, void* d_ws, size_t ws_size,
                              hipStream_t stream) {
  const float* x = (const float*)d_in[0];   // (8192,128)
  const float* D = (const float*)d_in[1];   // (1024,128)
  float* Z = (float*)d_out;                 // (8192,1024)

  char* ws = (char*)d_ws;
  size_t off = 0;
  const size_t BN = (size_t)B_ROWS * N_ATOMS;          // 8M elems
  float*  C  = (float*)(ws + off); off += BN * sizeof(float);              // 32MB
  float*  P0 = (float*)(ws + off); off += BN * sizeof(float);              // 32MB (Y)
  float*  P1 = (float*)(ws + off); off += BN * sizeof(float);              // 32MB (V)
  float*  G  = (float*)(ws + off); off += (size_t)N_ATOMS * N_ATOMS * 4;   // 4MB
  float*  Dn = (float*)(ws + off); off += (size_t)N_ATOMS * FDIM * 4;      // 512KB
  __bf16* Gp = (__bf16*)(ws + off); off += (size_t)N_ATOMS * N_ATOMS * 2;  // 2MB
  float*  stepPtr = (float*)(ws + off); off += 256;
  (void)in_sizes; (void)n_in; (void)out_size; (void)ws_size;

  // one-time setup (recomputed every call for determinism; inputs unchanged)
  k_norm_D<<<N_ATOMS, FDIM, 0, stream>>>(D, Dn);
  k_gram  <<<dim3(N_ATOMS / 16, N_ATOMS / 16), dim3(16, 16), 0, stream>>>(Dn, G);
  k_power <<<1, N_ATOMS, 0, stream>>>(G, stepPtr);
  k_pack_G<<<(N_ATOMS * N_ATOMS) / 256, 256, 0, stream>>>(G, Gp);
  k_xDT   <<<dim3(N_ATOMS / 16, B_ROWS / 16), dim3(16, 16), 0, stream>>>(x, Dn, C);
  k_zero2 <<<(unsigned)(BN / 256), 256, 0, stream>>>(Z, P0);   // Z0 = Y0 = 0

  float t = 1.0f;
  for (int it = 0; it < N_ITERS; ++it) {
    const float tn  = 0.5f * (1.0f + sqrtf(1.0f + 4.0f * t * t));
    const float mom = (t - 1.0f) / tn;
    k_fista_gemm<<<dim3(N_ATOMS / 256, B_ROWS / 32), 256, 0, stream>>>(P0, Gp, C, stepPtr, P1);
    k_topk      <<<B_ROWS, 256, 0, stream>>>(P1, Z, P0, mom);
    t = tn;
  }
}
